// QuantumGATLayer_39608188404039
// MI455X (gfx1250) — compile-verified, matrix operand order checked by reading, not run on hardware
//
#include <hip/hip_runtime.h>
#include <hip/hip_bf16.h>

#define IN_CH   128
#define HEADS   8
#define OUT_CH  16
#define OUT_DIM 128
#define NEG_SLOPE 0.2f
#define LN_EPS    1e-5f

typedef __attribute__((ext_vector_type(16))) _Float16 v16h;
typedef __attribute__((ext_vector_type(8)))  _Float16 v8h;
typedef __attribute__((ext_vector_type(8)))  float    v8f;
typedef __attribute__((ext_vector_type(4)))  float    v4f;

// ---------------------------------------------------------------------------
// Wt[n][k] = (f16) W[k][n]   (W is [IN_CH, OUT_DIM] row-major)
// ---------------------------------------------------------------------------
__global__ void k_prep_w(const float* __restrict__ W, _Float16* __restrict__ Wt) {
  int t = blockIdx.x * blockDim.x + threadIdx.x;   // 128*128 = 16384 threads
  int n = t >> 7, k = t & 127;
  Wt[n * IN_CH + k] = (_Float16)W[k * OUT_DIM + n];
}

// ---------------------------------------------------------------------------
// out = 0, emax = -inf, denom = 0
// ---------------------------------------------------------------------------
__global__ void k_init(float* __restrict__ out, float* __restrict__ emax,
                       float* __restrict__ denom, int N) {
  long long t = (long long)blockIdx.x * blockDim.x + threadIdx.x;
  long long tot = (long long)N * OUT_DIM;
  if (t < tot) out[t] = 0.f;
  if (t < (long long)N * HEADS) {
    emax[t]  = -__builtin_inff();
    denom[t] = 0.f;
  }
}

// ---------------------------------------------------------------------------
// h = x @ W via v_wmma_f32_16x16x32_f16. One wave per 16x16 output tile.
// lane%16 = row (A) / col (B), lane/16 = K-octet select; VGPR j/2 packs K pair.
// Requires N % 16 == 0 (N = 100000 -> 6250 row tiles).
// ---------------------------------------------------------------------------
__global__ void __launch_bounds__(256)
k_gemm(const float* __restrict__ x, const _Float16* __restrict__ Wt,
       float* __restrict__ h) {
  const int lane  = threadIdx.x & 31;
  const int wave  = threadIdx.x >> 5;        // 0..7 -> column tile
  const int lm    = lane & 15;
  const int kh    = lane >> 4;               // 0/1
  const int mbase = blockIdx.x * 16;
  const int nbase = wave * 16;

  const float*    xr = x  + (size_t)(mbase + lm) * IN_CH;
  const _Float16* wr = Wt + (size_t)(nbase + lm) * IN_CH;

  v8f acc = {};
#pragma unroll
  for (int k0 = 0; k0 < IN_CH; k0 += 32) {
    const int ka = k0 + kh * 8;
    // A: x row, two contiguous 8-float runs (ka..ka+7, ka+16..ka+23) -> f16
    v4f f0 = *(const v4f*)(xr + ka);
    v4f f1 = *(const v4f*)(xr + ka + 4);
    v4f f2 = *(const v4f*)(xr + ka + 16);
    v4f f3 = *(const v4f*)(xr + ka + 20);
    v16h A;
#pragma unroll
    for (int j = 0; j < 4; ++j) {
      A[j]      = (_Float16)f0[j];
      A[4 + j]  = (_Float16)f1[j];
      A[8 + j]  = (_Float16)f2[j];
      A[12 + j] = (_Float16)f3[j];
    }
    // B: Wt row (already f16), same K pattern
    v8h b0 = *(const v8h*)(wr + ka);
    v8h b1 = *(const v8h*)(wr + ka + 16);
    v16h B;
#pragma unroll
    for (int j = 0; j < 8; ++j) { B[j] = b0[j]; B[8 + j] = b1[j]; }

    acc = __builtin_amdgcn_wmma_f32_16x16x32_f16(
        /*neg_a=*/false, A, /*neg_b=*/false, B,
        /*c_mod=*/(short)0, acc, /*reuse_a=*/false, /*reuse_b=*/false);
  }
  // C/D layout: VGPR i -> M = i (lanes 0-15) or i+8 (lanes 16-31), N = lane%16
  const int n = nbase + lm;
#pragma unroll
  for (int i = 0; i < 8; ++i) {
    int m = mbase + kh * 8 + i;
    h[(size_t)m * OUT_DIM + n] = acc[i];
  }
}

// ---------------------------------------------------------------------------
// a_s[n,h] = <h[n,h,:], att_src[h,:]> ; same for a_d. One thread per (n,h).
// ---------------------------------------------------------------------------
__global__ void k_attn(const float* __restrict__ h,
                       const float* __restrict__ att_src,
                       const float* __restrict__ att_dst,
                       float* __restrict__ a_s, float* __restrict__ a_d, int N) {
  int t = blockIdx.x * blockDim.x + threadIdx.x;
  if (t >= N * HEADS) return;
  int hd = t & (HEADS - 1);
  const float* hp = h + (size_t)t * OUT_CH;         // (node*8+head)*16
  const float* as = att_src + hd * OUT_CH;
  const float* ad = att_dst + hd * OUT_CH;
  float s = 0.f, d = 0.f;
#pragma unroll
  for (int c = 0; c < OUT_CH; ++c) { float v = hp[c]; s += v * as[c]; d += v * ad[c]; }
  a_s[t] = s; a_d[t] = d;
}

// ---------------------------------------------------------------------------
// sign-aware float atomic max (init must be -inf written through this scheme)
// ---------------------------------------------------------------------------
__device__ inline void atomicMaxF(float* addr, float val) {
  if (val >= 0.f) atomicMax((int*)addr, __float_as_int(val));
  else            atomicMin((unsigned int*)addr, (unsigned int)__float_as_int(val));
}

__device__ inline void edge_sd(const long long* ei, long long e, int E, int& s, int& d) {
  if (e < (long long)E) { s = (int)ei[e]; d = (int)ei[(size_t)E + e]; }
  else                  { s = d = (int)(e - E); }   // self loops appended
}

// ---------------------------------------------------------------------------
// pass 1: emax[dst,h] = max leaky_relu(a_s[src,h] + a_d[dst,h])
// ---------------------------------------------------------------------------
__global__ void k_edge_max(const long long* __restrict__ ei,
                           const float* __restrict__ a_s,
                           const float* __restrict__ a_d,
                           float* __restrict__ emax, int E, int N) {
  long long t = (long long)blockIdx.x * blockDim.x + threadIdx.x;
  if (t >= (long long)(E + N) * HEADS) return;
  long long e = t >> 3; int hd = (int)(t & 7);
  int s, d; edge_sd(ei, e, E, s, d);
  float v = a_s[s * HEADS + hd] + a_d[d * HEADS + hd];
  v = v > 0.f ? v : NEG_SLOPE * v;
  atomicMaxF(&emax[d * HEADS + hd], v);
}

// ---------------------------------------------------------------------------
// pass 2: denom[dst,h] += exp(lr - emax[dst,h])
// ---------------------------------------------------------------------------
__global__ void k_edge_sum(const long long* __restrict__ ei,
                           const float* __restrict__ a_s,
                           const float* __restrict__ a_d,
                           const float* __restrict__ emax,
                           float* __restrict__ denom, int E, int N) {
  long long t = (long long)blockIdx.x * blockDim.x + threadIdx.x;
  if (t >= (long long)(E + N) * HEADS) return;
  long long e = t >> 3; int hd = (int)(t & 7);
  int s, d; edge_sd(ei, e, E, s, d);
  float v = a_s[s * HEADS + hd] + a_d[d * HEADS + hd];
  v = v > 0.f ? v : NEG_SLOPE * v;
  atomicAdd(&denom[d * HEADS + hd], expf(v - emax[d * HEADS + hd]));
}

// ---------------------------------------------------------------------------
// pass 3: out[dst,:] += alpha * h[src,:]. One wave per edge; lane owns 4
// channels (lane/4 = head). h is L2-resident (51 MB < 192 MB L2).
// ---------------------------------------------------------------------------
__global__ void __launch_bounds__(256)
k_edge_agg(const long long* __restrict__ ei,
           const float* __restrict__ a_s, const float* __restrict__ a_d,
           const float* __restrict__ emax, const float* __restrict__ denom,
           const float* __restrict__ h, float* __restrict__ out, int E, int N) {
  const int lane = threadIdx.x & 31;
  long long e = (long long)blockIdx.x * (blockDim.x >> 5) + (threadIdx.x >> 5);
  if (e >= (long long)(E + N)) return;
  int s, d; edge_sd(ei, e, E, s, d);
  const int hd = lane >> 2;                       // 4 lanes per head
  float v = a_s[s * HEADS + hd] + a_d[d * HEADS + hd];
  v = v > 0.f ? v : NEG_SLOPE * v;
  const float alpha = expf(v - emax[d * HEADS + hd]) / denom[d * HEADS + hd];
  v4f hv = *(const v4f*)(h + (size_t)s * OUT_DIM + lane * 4);
  float* op = out + (size_t)d * OUT_DIM + lane * 4;
  atomicAdd(op + 0, alpha * hv[0]);
  atomicAdd(op + 1, alpha * hv[1]);
  atomicAdd(op + 2, alpha * hv[2]);
  atomicAdd(op + 3, alpha * hv[3]);
}

// ---------------------------------------------------------------------------
// bias + LayerNorm + ELU, in place. One wave per node, lane owns 4 channels.
// ---------------------------------------------------------------------------
__global__ void __launch_bounds__(256)
k_ln_elu(float* __restrict__ out, const float* __restrict__ bias,
         const float* __restrict__ gamma, const float* __restrict__ beta, int N) {
  const int lane = threadIdx.x & 31;
  long long n = (long long)blockIdx.x * (blockDim.x >> 5) + (threadIdx.x >> 5);
  if (n >= N) return;
  float* row = out + (size_t)n * OUT_DIM;
  const int c = lane * 4;
  v4f v = *(const v4f*)(row + c);
  v4f b = *(const v4f*)(bias + c);
  v = v + b;
  float sum = v[0] + v[1] + v[2] + v[3];
  float sq  = v[0]*v[0] + v[1]*v[1] + v[2]*v[2] + v[3]*v[3];
#pragma unroll
  for (int m = 16; m >= 1; m >>= 1) {
    sum += __shfl_xor(sum, m, 32);
    sq  += __shfl_xor(sq,  m, 32);
  }
  const float mu  = sum * (1.f / OUT_DIM);
  const float var = sq * (1.f / OUT_DIM) - mu * mu;
  const float inv = rsqrtf(var + LN_EPS);
  v4f g  = *(const v4f*)(gamma + c);
  v4f be = *(const v4f*)(beta + c);
  v4f o;
#pragma unroll
  for (int j = 0; j < 4; ++j) {
    float t = (v[j] - mu) * inv * g[j] + be[j];
    o[j] = t > 0.f ? t : expf(t) - 1.f;     // ELU (alpha=1)
  }
  *(v4f*)(row + c) = o;
}

// ---------------------------------------------------------------------------
extern "C" void kernel_launch(void* const* d_in, const int* in_sizes, int n_in,
                              void* d_out, int out_size, void* d_ws, size_t ws_size,
                              hipStream_t stream) {
  const float*     x        = (const float*)d_in[0];
  const long long* ei       = (const long long*)d_in[1];   // int64 [2,E]
  const float*     W        = (const float*)d_in[2];
  const float*     att_src  = (const float*)d_in[3];
  const float*     att_dst  = (const float*)d_in[4];
  const float*     bias     = (const float*)d_in[5];
  const float*     ln_gamma = (const float*)d_in[6];
  const float*     ln_beta  = (const float*)d_in[7];
  float*           out      = (float*)d_out;

  const int N = in_sizes[0] / IN_CH;       // 100000
  const int E = in_sizes[1] / 2;           // 1600000

  // workspace layout
  float* h      = (float*)d_ws;                       // N*128 f32
  float* a_s    = h + (size_t)N * OUT_DIM;            // N*8
  float* a_d    = a_s + (size_t)N * HEADS;            // N*8
  float* emax   = a_d + (size_t)N * HEADS;            // N*8
  float* denom  = emax + (size_t)N * HEADS;           // N*8
  _Float16* Wt  = (_Float16*)(denom + (size_t)N * HEADS); // 128*128 f16

  const long long ET = (long long)(E + N) * HEADS;    // edge-head count

  k_prep_w<<<(IN_CH * OUT_DIM) / 256, 256, 0, stream>>>(W, Wt);
  {
    long long tot = (long long)N * OUT_DIM;
    k_init<<<(unsigned)((tot + 255) / 256), 256, 0, stream>>>(out, emax, denom, N);
  }
  k_gemm<<<N / 16, 256, 0, stream>>>(x, Wt, h);
  k_attn<<<(N * HEADS + 255) / 256, 256, 0, stream>>>(h, att_src, att_dst, a_s, a_d, N);
  k_edge_max<<<(unsigned)((ET + 255) / 256), 256, 0, stream>>>(ei, a_s, a_d, emax, E, N);
  k_edge_sum<<<(unsigned)((ET + 255) / 256), 256, 0, stream>>>(ei, a_s, a_d, emax, denom, E, N);
  k_edge_agg<<<(unsigned)(((long long)(E + N) + 7) / 8), 256, 0, stream>>>(
      ei, a_s, a_d, emax, denom, h, out, E, N);
  k_ln_elu<<<(unsigned)(((long long)N + 7) / 8), 256, 0, stream>>>(
      out, bias, ln_gamma, ln_beta, N);
}
